// Or_60696477827722
// MI455X (gfx1250) — compile-verified, matrix operand order checked by reading, not run on hardware
//
#include <hip/hip_runtime.h>
#include <hip/hip_bf16.h>

typedef __attribute__((ext_vector_type(16))) __bf16          v16bf;
typedef __attribute__((ext_vector_type(8)))  float           v8f;
typedef __attribute__((ext_vector_type(16))) unsigned short  u16x16;
typedef __attribute__((ext_vector_type(8)))  unsigned short  u16x8;

#define N_ITEMS   8192
#define D_IN      1024
#define E_OUT     1024
#define K_EXP     8
#define OUT_STRIDE (E_OUT + K_EXP)

#define TM 64       // rows per block tile
#define TN 128      // encoding cols per block tile
#define TK 32       // K chunk (bf16 WMMA depth)
#define NFRAG 4     // 16-col fragments per wave (waveN covers 64 cols)
#define BSTRIDE 40  // padded bf16 K-stride per column in LDS (80B, 16B aligned)

// ws layout (ints): [0..7] counts, [8..16] offsets(9), [17..24] cursor, [32..32+N) perm
#define WS_CNT  0
#define WS_OFF  8
#define WS_CUR  17
#define WS_PERM 32

__device__ __forceinline__ unsigned short bf16_rne(float f) {
    unsigned int u = __builtin_bit_cast(unsigned int, f);
    unsigned int r = u + 0x7FFFu + ((u >> 16) & 1u);
    return (unsigned short)(r >> 16);
}

__device__ __forceinline__ void split_bf16(float f, unsigned short& hi, unsigned short& lo) {
    hi = bf16_rne(f);
    float fh = __builtin_bit_cast(float, (unsigned int)hi << 16);
    lo = bf16_rne(f - fh);
}

__global__ void zero_ws(int* ws) {
    if (threadIdx.x < 32) ws[threadIdx.x] = 0;
}

__global__ void count_tail(const int* __restrict__ type_ids, int* __restrict__ ws,
                           float* __restrict__ out) {
    int n = blockIdx.x * blockDim.x + threadIdx.x;
    if (n >= N_ITEMS) return;
    int t = type_ids[n];
    atomicAdd(&ws[WS_CNT + t], 1);
    float* tail = out + (long)n * OUT_STRIDE + E_OUT;
#pragma unroll
    for (int j = 0; j < K_EXP; ++j) tail[j] = (j == t) ? 1.0f : 0.0f;
}

__global__ void scan_counts(int* ws) {
    if (threadIdx.x != 0) return;
    int acc = 0;
    ws[WS_OFF + 0] = 0;
#pragma unroll
    for (int k = 0; k < K_EXP; ++k) {
        acc += ws[WS_CNT + k];
        ws[WS_OFF + k + 1] = acc;
        ws[WS_CUR + k] = ws[WS_OFF + k];
    }
}

__global__ void scatter_perm(const int* __restrict__ type_ids, int* __restrict__ ws) {
    int n = blockIdx.x * blockDim.x + threadIdx.x;
    if (n >= N_ITEMS) return;
    int t = type_ids[n];
    int pos = atomicAdd(&ws[WS_CUR + t], 1);
    ws[WS_PERM + pos] = n;
}

__global__ __launch_bounds__(256) void moe_gemm(const float* __restrict__ x,
                                                const float* __restrict__ W,
                                                const float* __restrict__ bvec,
                                                const int* __restrict__ ws,
                                                float* __restrict__ out) {
    // Ping-pong LDS buffers: one barrier per K-step.
    __shared__ __align__(16) unsigned short Bh0[TN * BSTRIDE];
    __shared__ __align__(16) unsigned short Bl0[TN * BSTRIDE];
    __shared__ __align__(16) unsigned short Bh1[TN * BSTRIDE];
    __shared__ __align__(16) unsigned short Bl1[TN * BSTRIDE];

    const int* cnt  = ws + WS_CNT;
    const int* offs = ws + WS_OFF;
    const int* perm = ws + WS_PERM;

    // Map blockIdx.x -> (expert, local row tile)
    int rt = blockIdx.x;
    int expert = -1, local = 0, acc = 0;
#pragma unroll
    for (int k = 0; k < K_EXP; ++k) {
        int c = cnt[k];
        int t = (c + TM - 1) / TM;
        if (expert < 0 && rt < acc + t) { expert = k; local = rt - acc; }
        acc += t;
    }
    if (expert < 0) return;

    const int start   = offs[expert];
    const int count   = cnt[expert];
    const int rowBase = local * TM;            // row offset within expert group
    const int c0      = blockIdx.y * TN;       // encoding column offset

    const int tidx  = threadIdx.x;
    const int lane  = tidx & 31;
    const int wave  = tidx >> 5;
    const int waveM = wave & 3;                // 4 row sub-tiles of 16
    const int waveN = wave >> 2;               // 2 col sub-tiles of 64

    // A fragment source row for this lane (M = lane&15; this code's contiguous k order: hi lanes K+16)
    const int  mrow   = waveM * 16 + (lane & 15);
    const int  pA     = rowBase + mrow;
    const bool validA = (pA < count);
    const float* xrow = x + (long)(validA ? perm[start + pA] : 0) * D_IN;
    const int  kShift = (lane >> 4) * 16;      // 0 or 16

    const float* Wbase = W + (long)expert * D_IN * E_OUT + c0;

    v8f cAcc[NFRAG] = {};

    for (int it = 0; it < D_IN / TK; ++it) {
        const int k0 = it * TK;
        unsigned short* SH = (it & 1) ? Bh1 : Bh0;
        unsigned short* SL = (it & 1) ? Bl1 : Bl0;

        // Cooperatively stage B chunk [TK x TN] -> LDS transposed [col][K], split hi/lo bf16
#pragma unroll
        for (int q = tidx; q < (TK * TN / 4); q += 256) {
            int d  = q >> 5;          // 0..31 (K within chunk)
            int cc = (q & 31) * 4;    // 0..124 (col within tile)
            const float4 wv = *(const float4*)(Wbase + (long)(k0 + d) * E_OUT + cc);
            float f[4] = {wv.x, wv.y, wv.z, wv.w};
#pragma unroll
            for (int i = 0; i < 4; ++i) {
                unsigned short hi, lo;
                split_bf16(f[i], hi, lo);
                SH[(cc + i) * BSTRIDE + d] = hi;
                SL[(cc + i) * BSTRIDE + d] = lo;
            }
        }

        // A fragments (hi/lo) straight from global (K-contiguous rows)
        u16x16 ahu = {}, alu = {};
        if (validA) {
            const float4* ap = (const float4*)(xrow + k0 + kShift);
#pragma unroll
            for (int j = 0; j < 4; ++j) {
                float4 av = ap[j];
                float f[4] = {av.x, av.y, av.z, av.w};
#pragma unroll
                for (int i = 0; i < 4; ++i) {
                    unsigned short hi, lo;
                    split_bf16(f[i], hi, lo);
                    ahu[j * 4 + i] = hi;
                    alu[j * 4 + i] = lo;
                }
            }
            if (k0 + TK < D_IN)
                __builtin_prefetch(xrow + k0 + TK + kShift, 0, 0);  // global_prefetch_b8
        }
        v16bf aH = __builtin_bit_cast(v16bf, ahu);
        v16bf aL = __builtin_bit_cast(v16bf, alu);

        __syncthreads();   // stage(it) visible; reads of buffer (it-2)%2 already retired

#pragma unroll
        for (int fgi = 0; fgi < NFRAG; ++fgi) {
            const int col = waveN * 64 + fgi * 16 + (lane & 15);
            const u16x8* ph = (const u16x8*)&SH[col * BSTRIDE + kShift];
            const u16x8* pl = (const u16x8*)&SL[col * BSTRIDE + kShift];
            u16x8 h0 = ph[0], h1 = ph[1];
            u16x8 l0 = pl[0], l1 = pl[1];
            u16x16 bhu, blu;
#pragma unroll
            for (int i = 0; i < 8; ++i) {
                bhu[i] = h0[i]; bhu[i + 8] = h1[i];
                blu[i] = l0[i]; blu[i + 8] = l1[i];
            }
            v16bf bH = __builtin_bit_cast(v16bf, bhu);
            v16bf bL = __builtin_bit_cast(v16bf, blu);

            // split product: xh*Wh + xh*Wl + xl*Wh  (f32 accumulate)
            cAcc[fgi] = __builtin_amdgcn_wmma_f32_16x16x32_bf16(
                false, aH, false, bH, (short)0, cAcc[fgi], false, false);
            cAcc[fgi] = __builtin_amdgcn_wmma_f32_16x16x32_bf16(
                false, aH, false, bL, (short)0, cAcc[fgi], false, false);
            cAcc[fgi] = __builtin_amdgcn_wmma_f32_16x16x32_bf16(
                false, aL, false, bH, (short)0, cAcc[fgi], false, false);
        }
    }

    // Epilogue: C layout -> lane holds column (lane&15)+16*fgi, rows v + 8*(lane>=16)
    const int hiHalf = lane >> 4;
    int  rows[8];
    bool rvalid[8];
#pragma unroll
    for (int v = 0; v < 8; ++v) {
        int p = rowBase + waveM * 16 + v + hiHalf * 8;
        rvalid[v] = (p < count);
        rows[v]   = rvalid[v] ? perm[start + p] : 0;
    }
#pragma unroll
    for (int fgi = 0; fgi < NFRAG; ++fgi) {
        const int col  = c0 + waveN * 64 + fgi * 16 + (lane & 15);
        const float bb = bvec[(long)expert * E_OUT + col];
#pragma unroll
        for (int v = 0; v < 8; ++v) {
            if (rvalid[v])
                out[(long)rows[v] * OUT_STRIDE + col] = cAcc[fgi][v] + bb;
        }
    }
}

extern "C" void kernel_launch(void* const* d_in, const int* in_sizes, int n_in,
                              void* d_out, int out_size, void* d_ws, size_t ws_size,
                              hipStream_t stream) {
    const float* x   = (const float*)d_in[0];
    const int*   tid = (const int*)d_in[1];
    const float* W   = (const float*)d_in[2];
    const float* b   = (const float*)d_in[3];
    float* out = (float*)d_out;
    int*   ws  = (int*)d_ws;

    zero_ws<<<1, 32, 0, stream>>>(ws);
    count_tail<<<N_ITEMS / 256, 256, 0, stream>>>(tid, ws, out);
    scan_counts<<<1, 32, 0, stream>>>(ws);
    scatter_perm<<<N_ITEMS / 256, 256, 0, stream>>>(tid, ws);

    dim3 grid(N_ITEMS / TM + K_EXP, E_OUT / TN);
    moe_gemm<<<grid, 256, 0, stream>>>(x, W, b, ws, out);
}
